// MoDAVisionTransformer_89618787599010
// MI455X (gfx1250) — matrix operand order repaired and checked by reading, not compiled, not run on hardware
//
#include <hip/hip_runtime.h>

typedef unsigned short ushort_t;
typedef unsigned int   uint_t;

typedef __attribute__((ext_vector_type(16))) __bf16 v16bf;
typedef __attribute__((ext_vector_type(8)))  float  v8f;

#define NB     8
#define N_TOK  197
#define T_TOK  1576          // NB * N_TOK
#define DIMQ   768
#define MLPQ   3072
#define HDQ    64
#define NGQ    2364          // N_TOK * 12 heads (query rows per batch)
#define KMAXP  2368          // padded max keys (12*197 -> mult of 32)
#define DEPTHQ 12

__device__ __forceinline__ ushort_t bf16_rne(float f) {
  uint_t u = __float_as_uint(f);
  u += 0x7FFFu + ((u >> 16) & 1u);
  return (ushort_t)(u >> 16);
}
__device__ __forceinline__ int imin(int a, int b) { return a < b ? a : b; }

union Frag16 { v16bf v; uint4 q[2]; };

enum { GF_BIAS = 1, GF_RESID = 2, GF_GELU = 4, GF_F32 = 8, GF_BF16 = 16 };

// ---------------------------------------------------------------------------
// Generic bf16 WMMA GEMM: C[M,N] = A[M,K] @ W[K,N] (+bias)(+GELU)(+resid)
// A: bf16 [M][K] row-major.  Wt: bf16 [Npad][K] (pre-transposed weights).
// 4 waves/WG cover a 64x64 tile (wave w = rows 16w..16w+15, all 64 cols).
// The weight tile (shared by all 4 waves) is staged into LDS once per K-step
// with gfx1250 async-to-LDS loads (ASYNCcnt), overlapping the per-wave
// A-fragment global loads; WMMA B-fragments are then fed from LDS.
// ---------------------------------------------------------------------------
__global__ __launch_bounds__(128) void gemm_bf16_wmma(
    const ushort_t* __restrict__ A, const ushort_t* __restrict__ Wt,
    const float* __restrict__ bias, float* __restrict__ Cf,
    ushort_t* __restrict__ Cb, int M, int N, int K, int ldc, int flags)
{
  __shared__ __align__(16) ushort_t Wlds[64 * 32];   // 64 cols x 32 K bf16 = 4KB
  const int wave = threadIdx.x >> 5;
  const int lane = threadIdx.x & 31;
  const int lh = lane >> 4, ll = lane & 15;
  const int row0 = blockIdx.y * 64 + wave * 16;
  const int col0 = blockIdx.x * 64;
  const int arow = imin(row0 + ll, M - 1);       // clamp partial M tiles
  const ushort_t* Ar = A + (size_t)arow * K;
  const unsigned ldsBase = (unsigned)(size_t)(&Wlds[0]);

  v8f acc[4] = {};
  for (int kk = 0; kk < K; kk += 32) {
    // ---- async-stage weight tile: 256 x 16B chunks, 2 per thread ----
    // chunk c2 -> LDS offset c2*16 ; global = Wt[(col0 + c2/4)*K + kk + (c2%4)*8]
#pragma unroll
    for (int i = 0; i < 2; ++i) {
      const int c2 = threadIdx.x + 128 * i;
      const unsigned long long gaddr = (unsigned long long)(size_t)(
          Wt + (size_t)(col0 + (c2 >> 2)) * K + kk + (c2 & 3) * 8);
      const unsigned loff = ldsBase + (unsigned)c2 * 16u;
      asm volatile("global_load_async_to_lds_b128 %0, %1, off"
                   :: "v"(loff), "v"(gaddr) : "memory");
    }
    if (kk + 64 < K) __builtin_prefetch((const void*)(Ar + kk + 64), 0, 3);
    // A fragment (16x32 bf16): lane holds row ll; K = (e/8)*16 + lh*8 + e%8
    Frag16 af;
    af.q[0] = *(const uint4*)(Ar + kk + lh * 8);
    af.q[1] = *(const uint4*)(Ar + kk + 16 + lh * 8);
    asm volatile("s_wait_asynccnt 0x0" ::: "memory");
    __syncthreads();                               // all waves' tile in LDS
#pragma unroll
    for (int t = 0; t < 4; ++t) {
      // B fragment (32x16): lane = col n (ll), K = (e/8)*16 + lh*8 + e%8
      Frag16 bfg;
      const ushort_t* wp = &Wlds[(t * 16 + ll) * 32 + lh * 8];
      bfg.q[0] = *(const uint4*)wp;
      bfg.q[1] = *(const uint4*)(wp + 16);
      acc[t] = __builtin_amdgcn_wmma_f32_16x16x32_bf16(
          false, af.v, false, bfg.v, (short)0, acc[t], false, false);
    }
    __syncthreads();                               // protect LDS for next step
  }
  // Epilogue. C/D layout: vgpr r, lane l -> (m = r + 8*lh, n = ll) within tile.
#pragma unroll
  for (int t = 0; t < 4; ++t) {
#pragma unroll
    for (int r = 0; r < 8; ++r) {
      const int m = row0 + r + 8 * lh;
      const int n = col0 + t * 16 + ll;
      if (m < M && n < N) {
        float v = acc[t][r];
        if (flags & GF_BIAS) v += bias[n];
        if (flags & GF_GELU) v = 0.5f * v * (1.0f + erff(v * 0.70710678f));
        const size_t idx = (size_t)m * ldc + n;
        if (flags & GF_RESID) v += Cf[idx];
        if (flags & GF_F32)  Cf[idx] = v;
        if (flags & GF_BF16) Cb[idx] = bf16_rne(v);
      }
    }
  }
}

// ---------------------------------------------------------------------------
// Flash attention, multi-query (1 KV head), keys accumulated over layers.
// One wave per (batch, 16-query tile). Streams 32-key blocks:
//   S = Q K^T (4 wmma), online softmax, O += P V (4 wmma, P re-laid via LDS).
// ---------------------------------------------------------------------------
__global__ __launch_bounds__(32) void attn_kernel(
    const ushort_t* __restrict__ Q,   // [B][NGQ][64] bf16
    const ushort_t* __restrict__ Kc,  // [B][KMAXP][64] bf16
    const ushort_t* __restrict__ Vt,  // [B][64][KMAXP] bf16 (transposed V)
    ushort_t* __restrict__ O,         // [B][NGQ][64] bf16
    int Lk)
{
  __shared__ __align__(16) ushort_t Plds[16 * 32];
  const int b  = blockIdx.y;
  const int qt = blockIdx.x;
  const int lane = threadIdx.x;
  const int lh = lane >> 4, ll = lane & 15;
  const ushort_t* Qb = Q  + (size_t)b * NGQ   * HDQ;
  const ushort_t* Kb = Kc + (size_t)b * KMAXP * HDQ;
  const ushort_t* Vb = Vt + (size_t)b * HDQ   * KMAXP;

  const int qrow = imin(qt * 16 + ll, NGQ - 1);
  Frag16 qa[2];                                   // Q tile 16x64: two K=32 chunks
#pragma unroll
  for (int c = 0; c < 2; ++c) {
    const ushort_t* qp = Qb + (size_t)qrow * HDQ + c * 32 + lh * 8;
    qa[c].q[0] = *(const uint4*)qp;
    qa[c].q[1] = *(const uint4*)(qp + 16);
  }
  v8f ot[4] = {};
  float mrow[8], srow[8];
#pragma unroll
  for (int r = 0; r < 8; ++r) { mrow[r] = -1e30f; srow[r] = 0.f; }

  const int nkb = (Lk + 31) / 32;
  for (int kb0 = 0; kb0 < nkb; ++kb0) {
    const int kb = kb0 * 32;
    v8f s[2] = {};
#pragma unroll
    for (int j = 0; j < 2; ++j) {                 // two 16-key score tiles
      const int krow = imin(kb + j * 16 + ll, KMAXP - 1);
#pragma unroll
      for (int c = 0; c < 2; ++c) {               // head-dim K=64 in two chunks
        Frag16 kf;
        const ushort_t* kp = Kb + (size_t)krow * HDQ + c * 32 + lh * 8;
        kf.q[0] = *(const uint4*)kp;
        kf.q[1] = *(const uint4*)(kp + 16);
        s[j] = __builtin_amdgcn_wmma_f32_16x16x32_bf16(
            false, qa[c].v, false, kf.v, (short)0, s[j], false, false);
      }
    }
    // scale + mask out-of-range keys (col = ll in C layout)
#pragma unroll
    for (int j = 0; j < 2; ++j) {
      const bool valid = (kb + j * 16 + ll) < Lk;
#pragma unroll
      for (int r = 0; r < 8; ++r)
        s[j][r] = valid ? s[j][r] * 0.125f : -1e30f;
    }
    // online softmax: half-wave (16-lane) butterfly reductions per row
#pragma unroll
    for (int r = 0; r < 8; ++r) {
      float bm = fmaxf(s[0][r], s[1][r]);
#pragma unroll
      for (int mm = 8; mm >= 1; mm >>= 1) bm = fmaxf(bm, __shfl_xor(bm, mm, 32));
      const float nm = fmaxf(mrow[r], bm);
      const float fs = __expf(mrow[r] - nm);
      mrow[r] = nm;
      const float p0 = __expf(s[0][r] - nm);
      const float p1 = __expf(s[1][r] - nm);
      float ps = p0 + p1;
#pragma unroll
      for (int mm = 8; mm >= 1; mm >>= 1) ps += __shfl_xor(ps, mm, 32);
      srow[r] = srow[r] * fs + ps;
#pragma unroll
      for (int t = 0; t < 4; ++t) ot[t][r] *= fs;
      s[0][r] = p0; s[1][r] = p1;
    }
    // P: C-layout -> LDS -> A-fragment layout (16x32 bf16)
#pragma unroll
    for (int r = 0; r < 8; ++r) {
      const int row = r + 8 * lh;
      Plds[row * 32 + ll]      = bf16_rne(s[0][r]);
      Plds[row * 32 + 16 + ll] = bf16_rne(s[1][r]);
    }
    __syncthreads();
    Frag16 pf;
    pf.q[0] = *(const uint4*)&Plds[ll * 32 + lh * 8];
    pf.q[1] = *(const uint4*)&Plds[ll * 32 + 16 + lh * 8];
    __syncthreads();
    // O += P @ V   (V fragments contiguous thanks to transposed cache)
#pragma unroll
    for (int t = 0; t < 4; ++t) {
      Frag16 vf;
      const ushort_t* vp = Vb + (size_t)(t * 16 + ll) * KMAXP + kb + lh * 8;
      vf.q[0] = *(const uint4*)vp;
      vf.q[1] = *(const uint4*)(vp + 16);
      ot[t] = __builtin_amdgcn_wmma_f32_16x16x32_bf16(
          false, pf.v, false, vf.v, (short)0, ot[t], false, false);
    }
  }
#pragma unroll
  for (int t = 0; t < 4; ++t) {
#pragma unroll
    for (int r = 0; r < 8; ++r) {
      const int m = qt * 16 + r + 8 * lh;
      if (m < NGQ)
        O[((size_t)b * NGQ + m) * HDQ + t * 16 + ll] = bf16_rne(ot[t][r] / srow[r]);
    }
  }
}

// ---------------------------------------------------------------------------
// Small helper kernels
// ---------------------------------------------------------------------------
__global__ void wconv_kernel(const float* __restrict__ W, ushort_t* __restrict__ Wt,
                             int K, int N, int Npad)
{ // Wt[n][k] = bf16(W[k][n]); zero-pad n >= N
  const size_t idx = (size_t)blockIdx.x * blockDim.x + threadIdx.x;
  if (idx >= (size_t)Npad * K) return;
  const int k = (int)(idx % K);
  const int n = (int)(idx / K);
  Wt[idx] = bf16_rne(n < N ? W[(size_t)k * N + n] : 0.f);
}

__global__ void patchify_kernel(const float* __restrict__ x, ushort_t* __restrict__ Xp)
{ // Xp[b*196 + gy*14+gx][c*256 + i*16 + j] = x[b][c][gy*16+i][gx*16+j]
  const int idx = blockIdx.x * blockDim.x + threadIdx.x;
  if (idx >= 1568 * DIMQ) return;
  const int col = idx % DIMQ, rowp = idx / DIMQ;
  const int b = rowp / 196, pp = rowp % 196;
  const int gy = pp / 14, gx = pp % 14;
  const int c = col >> 8, ij = col & 255, i = ij >> 4, j = ij & 15;
  const float v = x[(((size_t)b * 3 + c) * 224 + gy * 16 + i) * 224 + gx * 16 + j];
  Xp[idx] = bf16_rne(v);
}

__global__ void assemble_h_kernel(const float* __restrict__ tok,
                                  const float* __restrict__ cls,
                                  const float* __restrict__ pos,
                                  float* __restrict__ h)
{
  const int idx = blockIdx.x * blockDim.x + threadIdx.x;
  if (idx >= T_TOK * DIMQ) return;
  const int d = idx % DIMQ, t = idx / DIMQ;
  const int b = t / N_TOK, n = t % N_TOK;
  const float v = (n == 0) ? cls[d] : tok[((size_t)b * 196 + (n - 1)) * DIMQ + d];
  h[idx] = v + pos[(size_t)n * DIMQ + d];
}

__global__ __launch_bounds__(32) void ln_rows_kernel(
    const float* __restrict__ X, const float* __restrict__ w,
    const float* __restrict__ bb, ushort_t* __restrict__ Y, int Trows)
{ // one wave per row of 768
  const int row = blockIdx.x;
  if (row >= Trows) return;
  const float* xr = X + (size_t)row * DIMQ;
  float v[24];
  float s = 0.f;
#pragma unroll
  for (int i = 0; i < 24; ++i) { v[i] = xr[threadIdx.x + i * 32]; s += v[i]; }
#pragma unroll
  for (int mm = 16; mm >= 1; mm >>= 1) s += __shfl_xor(s, mm, 32);
  const float mean = s * (1.f / 768.f);
  float q = 0.f;
#pragma unroll
  for (int i = 0; i < 24; ++i) { const float d = v[i] - mean; q += d * d; }
#pragma unroll
  for (int mm = 16; mm >= 1; mm >>= 1) q += __shfl_xor(q, mm, 32);
  const float inv = rsqrtf(q * (1.f / 768.f) + 1e-6f);
#pragma unroll
  for (int i = 0; i < 24; ++i) {
    const int col = threadIdx.x + i * 32;
    Y[(size_t)row * DIMQ + col] = bf16_rne((v[i] - mean) * inv * w[col] + bb[col]);
  }
}

__global__ void kv_scatter_kernel(const float* __restrict__ kvf,
                                  ushort_t* __restrict__ Kc,
                                  ushort_t* __restrict__ Vt, int layer)
{
  const int idx = blockIdx.x * blockDim.x + threadIdx.x;
  if (idx >= T_TOK * HDQ) return;
  const int d = idx & 63, t = idx >> 6;
  const int b = t / N_TOK, n = t % N_TOK;
  const int j = layer * N_TOK + n;
  Kc[((size_t)b * KMAXP + j) * HDQ + d] = bf16_rne(kvf[(size_t)t * 128 + d]);
  Vt[((size_t)b * HDQ + d) * KMAXP + j] = bf16_rne(kvf[(size_t)t * 128 + 64 + d]);
}

__global__ void extract_cls_kernel(const ushort_t* __restrict__ hn,
                                   ushort_t* __restrict__ hc)
{
  const int idx = blockIdx.x * blockDim.x + threadIdx.x;
  if (idx >= NB * DIMQ) return;
  const int b = idx / DIMQ, d = idx % DIMQ;
  hc[idx] = hn[(size_t)(b * N_TOK) * DIMQ + d];
}

// ---------------------------------------------------------------------------
extern "C" void kernel_launch(void* const* d_in, const int* in_sizes, int n_in,
                              void* d_out, int out_size, void* d_ws, size_t ws_size,
                              hipStream_t stream) {
  (void)in_sizes; (void)n_in; (void)out_size; (void)ws_size;
  const float* x       = (const float*)d_in[0];
  const float* patch_w = (const float*)d_in[1];
  const float* patch_b = (const float*)d_in[2];
  const float* cls_tok = (const float*)d_in[3];
  const float* pos_emb = (const float*)d_in[4];
  const float* ln1_w   = (const float*)d_in[5];
  const float* ln1_b   = (const float*)d_in[6];
  const float* q_w     = (const float*)d_in[7];
  const float* q_b     = (const float*)d_in[8];
  const float* kv_w    = (const float*)d_in[9];
  const float* kv_b    = (const float*)d_in[10];
  const float* proj_w  = (const float*)d_in[11];
  const float* proj_b  = (const float*)d_in[12];
  const float* ln2_w   = (const float*)d_in[13];
  const float* ln2_b   = (const float*)d_in[14];
  const float* fc1_w   = (const float*)d_in[15];
  const float* fc1_b   = (const float*)d_in[16];
  const float* fc2_w   = (const float*)d_in[17];
  const float* fc2_b   = (const float*)d_in[18];
  const float* norm_w  = (const float*)d_in[19];
  const float* norm_b  = (const float*)d_in[20];
  const float* head_w  = (const float*)d_in[21];
  const float* head_b  = (const float*)d_in[22];
  float* out = (float*)d_out;

  // carve workspace (stream-serialized, so one weight slot is reused)
  uint8_t* p = (uint8_t*)d_ws;
  auto carve = [&](size_t bytes) -> void* {
    uint8_t* r = p; p += (bytes + 255) & ~(size_t)255; return (void*)r;
  };
  ushort_t* WT  = (ushort_t*)carve((size_t)MLPQ * DIMQ * 2);   // weight slot (max fc1)
  ushort_t* XP  = (ushort_t*)carve((size_t)1568 * DIMQ * 2);   // patches bf16
  float*    TOK = (float*)   carve((size_t)1568 * DIMQ * 4);   // patch embed f32
  float*    H   = (float*)   carve((size_t)T_TOK * DIMQ * 4);  // residual stream f32
  ushort_t* HN  = (ushort_t*)carve((size_t)T_TOK * DIMQ * 2);  // LN output bf16
  ushort_t* QB  = (ushort_t*)carve((size_t)T_TOK * DIMQ * 2);  // Q bf16 (== [B][2364][64])
  float*    KVF = (float*)   carve((size_t)T_TOK * 128 * 4);   // kv f32
  ushort_t* KC  = (ushort_t*)carve((size_t)NB * KMAXP * HDQ * 2); // K cache
  ushort_t* VT  = (ushort_t*)carve((size_t)NB * HDQ * KMAXP * 2); // V cache (transposed)
  ushort_t* OB  = (ushort_t*)carve((size_t)T_TOK * DIMQ * 2);  // attn out bf16
  ushort_t* M1  = (ushort_t*)carve((size_t)T_TOK * MLPQ * 2);  // MLP hidden bf16
  ushort_t* HC  = (ushort_t*)carve((size_t)NB * DIMQ * 2);     // cls rows bf16

  // ---- patch embedding ----
  patchify_kernel<<<(1568 * DIMQ + 255) / 256, 256, 0, stream>>>(x, XP);
  wconv_kernel<<<(DIMQ * DIMQ + 255) / 256, 256, 0, stream>>>(patch_w, WT, DIMQ, DIMQ, DIMQ);
  {
    dim3 g(DIMQ / 64, (1568 + 63) / 64);
    gemm_bf16_wmma<<<g, 128, 0, stream>>>(XP, WT, patch_b, TOK, nullptr,
                                          1568, DIMQ, DIMQ, DIMQ, GF_BIAS | GF_F32);
  }
  assemble_h_kernel<<<(T_TOK * DIMQ + 255) / 256, 256, 0, stream>>>(TOK, cls_tok, pos_emb, H);

  // ---- transformer layers ----
  for (int l = 0; l < DEPTHQ; ++l) {
    ln_rows_kernel<<<T_TOK, 32, 0, stream>>>(H, ln1_w + l * DIMQ, ln1_b + l * DIMQ, HN, T_TOK);

    wconv_kernel<<<(DIMQ * DIMQ + 255) / 256, 256, 0, stream>>>(
        q_w + (size_t)l * DIMQ * DIMQ, WT, DIMQ, DIMQ, DIMQ);
    { dim3 g(DIMQ / 64, (T_TOK + 63) / 64);
      gemm_bf16_wmma<<<g, 128, 0, stream>>>(HN, WT, q_b + l * DIMQ, nullptr, QB,
                                            T_TOK, DIMQ, DIMQ, DIMQ, GF_BIAS | GF_BF16); }

    wconv_kernel<<<(DIMQ * 128 + 255) / 256, 256, 0, stream>>>(
        kv_w + (size_t)l * DIMQ * 128, WT, DIMQ, 128, 128);
    { dim3 g(128 / 64, (T_TOK + 63) / 64);
      gemm_bf16_wmma<<<g, 128, 0, stream>>>(HN, WT, kv_b + l * 128, KVF, nullptr,
                                            T_TOK, 128, DIMQ, 128, GF_BIAS | GF_F32); }
    kv_scatter_kernel<<<(T_TOK * HDQ + 255) / 256, 256, 0, stream>>>(KVF, KC, VT, l);

    { dim3 g(148, NB);  // ceil(2364/16) query tiles x batch
      attn_kernel<<<g, 32, 0, stream>>>(QB, KC, VT, OB, (l + 1) * N_TOK); }

    wconv_kernel<<<(DIMQ * DIMQ + 255) / 256, 256, 0, stream>>>(
        proj_w + (size_t)l * DIMQ * DIMQ, WT, DIMQ, DIMQ, DIMQ);
    { dim3 g(DIMQ / 64, (T_TOK + 63) / 64);
      gemm_bf16_wmma<<<g, 128, 0, stream>>>(OB, WT, proj_b + l * DIMQ, H, nullptr,
                                            T_TOK, DIMQ, DIMQ, DIMQ,
                                            GF_BIAS | GF_RESID | GF_F32); }

    ln_rows_kernel<<<T_TOK, 32, 0, stream>>>(H, ln2_w + l * DIMQ, ln2_b + l * DIMQ, HN, T_TOK);

    wconv_kernel<<<(DIMQ * MLPQ + 255) / 256, 256, 0, stream>>>(
        fc1_w + (size_t)l * DIMQ * MLPQ, WT, DIMQ, MLPQ, MLPQ);
    { dim3 g(MLPQ / 64, (T_TOK + 63) / 64);
      gemm_bf16_wmma<<<g, 128, 0, stream>>>(HN, WT, fc1_b + l * MLPQ, nullptr, M1,
                                            T_TOK, MLPQ, DIMQ, MLPQ,
                                            GF_BIAS | GF_GELU | GF_BF16); }

    wconv_kernel<<<(MLPQ * DIMQ + 255) / 256, 256, 0, stream>>>(
        fc2_w + (size_t)l * MLPQ * DIMQ, WT, MLPQ, DIMQ, DIMQ);
    { dim3 g(DIMQ / 64, (T_TOK + 63) / 64);
      gemm_bf16_wmma<<<g, 128, 0, stream>>>(M1, WT, fc2_b + l * DIMQ, H, nullptr,
                                            T_TOK, DIMQ, MLPQ, DIMQ,
                                            GF_BIAS | GF_RESID | GF_F32); }
  }

  // ---- final norm + head ----
  ln_rows_kernel<<<T_TOK, 32, 0, stream>>>(H, norm_w, norm_b, HN, T_TOK);
  extract_cls_kernel<<<(NB * DIMQ + 255) / 256, 256, 0, stream>>>(HN, HC);
  wconv_kernel<<<(1024 * DIMQ + 255) / 256, 256, 0, stream>>>(head_w, WT, DIMQ, 1000, 1024);
  { dim3 g((1000 + 63) / 64, 1);
    gemm_bf16_wmma<<<g, 128, 0, stream>>>(HC, WT, head_b, out, nullptr,
                                          NB, 1000, DIMQ, 1000, GF_BIAS | GF_F32); }
}
